// MultiHeadAttention_74028056314029
// MI455X (gfx1250) — matrix operand order repaired and checked by reading, not compiled, not run on hardware
//
#include <hip/hip_runtime.h>

typedef __attribute__((ext_vector_type(16))) __bf16        v16bf;
typedef __attribute__((ext_vector_type(8)))  float         v8f;
typedef __attribute__((ext_vector_type(4)))  unsigned int  v4u;
typedef unsigned short u16;

#define Cdim 2048   // sequence length
#define Edim 2048   // embed dim
#define Hn   32     // query heads
#define KVn  8      // kv heads
#define Dh   64     // head dim
#define HD   2048   // Hn*Dh
#define KVD  512    // KVn*Dh

__device__ __forceinline__ u16 f2bf(float f) {
  unsigned int u = __float_as_uint(f);
  u += 0x7FFFu + ((u >> 16) & 1u);          // round-to-nearest-even
  return (u16)(u >> 16);
}

// Low 32 bits of a generic pointer to __shared__ = wave-relative LDS offset
__device__ __forceinline__ unsigned lds_off32(const void* p) {
  return (unsigned)(unsigned long long)p;
}

// CDNA5 async global->LDS copy, 16 bytes per lane (ASYNCcnt-tracked)
__device__ __forceinline__ void async_copy_b128(unsigned lds, const void* g) {
  asm volatile("global_load_async_to_lds_b128 %0, %1, off"
               :: "v"(lds), "v"((unsigned long long)g) : "memory");
}
__device__ __forceinline__ void wait_async0() {
  asm volatile("s_wait_asynccnt 0" ::: "memory");
}

union Frag { v16bf v; v4u q[2]; };

// ---------------------------------------------------------------- conversions
__global__ void f32_to_bf16_kernel(const float* __restrict__ in,
                                   u16* __restrict__ out, int n) {
  int i = blockIdx.x * blockDim.x + threadIdx.x;
  if (i < n) out[i] = f2bf(in[i]);
}

// V: [C][KVD] -> Vt: [KVD][C]  (so PV B-fragments are contiguous loads)
__global__ void transpose_v_kernel(const u16* __restrict__ V,
                                   u16* __restrict__ Vt) {
  int i = blockIdx.x * blockDim.x + threadIdx.x;   // over C*KVD
  int s = i / KVD, c = i % KVD;
  Vt[(size_t)c * Cdim + s] = V[i];
}

// ---------------------------------------------------------------- GEMM
// C[M,N] = A[M,K](bf16) * W[N,K](bf16)^T, f32 accumulate.
// Block: 256 threads = 8 waves; block tile 128x64; wave tile 32x32
// (2x2 register blocking -> 4 WMMAs per K-step, A/B frags reused twice).
// LDS tiles are filled with GLOBAL_LOAD_ASYNC_TO_LDS_B128.
template <bool OUT_F32>
__global__ void gemm_bf16_kernel(const u16* __restrict__ A,
                                 const u16* __restrict__ W,
                                 void* __restrict__ Cout, int N, int K) {
  __shared__ __align__(16) u16 As[128 * 32];  // [m][k]  8 KiB
  __shared__ __align__(16) u16 Ws[64 * 32];   // [n][k]  4 KiB
  const int tid  = threadIdx.x;
  const int lane = tid & 31, w = tid >> 5;
  const int wm = w >> 1, wn = w & 1;          // 4x2 wave grid
  const int r = lane & 15, hi = lane >> 4;
  const int m0 = blockIdx.y * 128, n0 = blockIdx.x * 64;
  const int arow = tid >> 2, ak = (tid & 3) * 8;   // A: 2 chunks of 8 halfs
  const int wrow = tid >> 2, wk = (tid & 3) * 8;   // W: 1 chunk of 8 halfs
  const unsigned ldsA0 = lds_off32(&As[arow * 32 + ak]);
  const unsigned ldsA1 = lds_off32(&As[(64 + arow) * 32 + ak]);
  const unsigned ldsW  = lds_off32(&Ws[wrow * 32 + wk]);
  v8f acc00 = {}, acc01 = {}, acc10 = {}, acc11 = {};
  for (int k0 = 0; k0 < K; k0 += 32) {
    async_copy_b128(ldsA0, &A[(size_t)(m0 + arow) * K + k0 + ak]);
    async_copy_b128(ldsA1, &A[(size_t)(m0 + 64 + arow) * K + k0 + ak]);
    async_copy_b128(ldsW,  &W[(size_t)(n0 + wrow) * K + k0 + wk]);
    wait_async0();
    __syncthreads();
    Frag fa0, fa1, fb0, fb1;
    // A frags: lane row = wave m-tile + r; k chunks at hi*8 and 16+hi*8
    fa0.q[0] = *(const v4u*)&As[(wm * 32 + r) * 32 + hi * 8];
    fa0.q[1] = *(const v4u*)&As[(wm * 32 + r) * 32 + 16 + hi * 8];
    fa1.q[0] = *(const v4u*)&As[(wm * 32 + 16 + r) * 32 + hi * 8];
    fa1.q[1] = *(const v4u*)&As[(wm * 32 + 16 + r) * 32 + 16 + hi * 8];
    // B frags: lane col = wave n-tile + r; k chunks at hi*8 and 16+hi*8
    fb0.q[0] = *(const v4u*)&Ws[(wn * 32 + r) * 32 + hi * 8];
    fb0.q[1] = *(const v4u*)&Ws[(wn * 32 + r) * 32 + 16 + hi * 8];
    fb1.q[0] = *(const v4u*)&Ws[(wn * 32 + 16 + r) * 32 + hi * 8];
    fb1.q[1] = *(const v4u*)&Ws[(wn * 32 + 16 + r) * 32 + 16 + hi * 8];
    acc00 = __builtin_amdgcn_wmma_f32_16x16x32_bf16(false, fa0.v, false, fb0.v, (short)0, acc00, false, false);
    acc01 = __builtin_amdgcn_wmma_f32_16x16x32_bf16(false, fa0.v, false, fb1.v, (short)0, acc01, false, false);
    acc10 = __builtin_amdgcn_wmma_f32_16x16x32_bf16(false, fa1.v, false, fb0.v, (short)0, acc10, false, false);
    acc11 = __builtin_amdgcn_wmma_f32_16x16x32_bf16(false, fa1.v, false, fb1.v, (short)0, acc11, false, false);
    __syncthreads();
  }
  const int gm0 = m0 + wm * 32 + hi * 8;      // rows gm0..gm0+7 (C/D layout)
  const int gn0 = n0 + wn * 32 + r;
#pragma unroll
  for (int v = 0; v < 8; ++v) {
    if constexpr (OUT_F32) {
      float* C = (float*)Cout;
      C[(size_t)(gm0 + v) * N + gn0]           = acc00[v];
      C[(size_t)(gm0 + v) * N + gn0 + 16]      = acc01[v];
      C[(size_t)(gm0 + 16 + v) * N + gn0]      = acc10[v];
      C[(size_t)(gm0 + 16 + v) * N + gn0 + 16] = acc11[v];
    } else {
      u16* C = (u16*)Cout;
      C[(size_t)(gm0 + v) * N + gn0]           = f2bf(acc00[v]);
      C[(size_t)(gm0 + v) * N + gn0 + 16]      = f2bf(acc01[v]);
      C[(size_t)(gm0 + 16 + v) * N + gn0]      = f2bf(acc10[v]);
      C[(size_t)(gm0 + 16 + v) * N + gn0 + 16] = f2bf(acc11[v]);
    }
  }
}

// ---------------------------------------------------------------- attention
// One wave per (16-row q tile, head). Flash-style online softmax.
__global__ void gqa_attention_kernel(const u16* __restrict__ Q,
                                     const u16* __restrict__ Km,
                                     const u16* __restrict__ Vt,
                                     u16* __restrict__ Hid) {
  __shared__ __align__(16) u16 Pl[16 * 32];   // P staging (A-frag transpose)
  const int lane = threadIdx.x;
  const int r = lane & 15, hi = lane >> 4;
  const int qt = blockIdx.x, h = blockIdx.y;
  const int kv = h >> 2;                      // GQA: 4 query heads per kv head
  const int q0 = qt * 16;
  const float scale = 0.125f;                 // 1/sqrt(64)

  // Q A-fragments for d-blocks [0,32) and [32,64) — constant over the loop
  Frag qa0, qa1;
  const u16* qp = Q + (size_t)(q0 + r) * HD + h * Dh;
  qa0.q[0] = *(const v4u*)&qp[hi * 8];
  qa0.q[1] = *(const v4u*)&qp[16 + hi * 8];
  qa1.q[0] = *(const v4u*)&qp[32 + hi * 8];
  qa1.q[1] = *(const v4u*)&qp[48 + hi * 8];

  v8f o0 = {}, o1 = {}, o2 = {}, o3 = {};     // 16x64 output accumulator
  float mrun[8], lrun[8];
#pragma unroll
  for (int v = 0; v < 8; ++v) { mrun[v] = -1e30f; lrun[v] = 0.f; }

  for (int sb = 0; sb < q0 + 16; sb += 32) {
    // ---- scores S = Q K^T for s in [sb, sb+32): two 16x16 tiles, K-dim 64
    Frag kb;
    v8f s0 = {}, s1 = {};
    const u16* kp0 = Km + (size_t)(sb + r) * KVD + kv * Dh;         // tile 0
    kb.q[0] = *(const v4u*)&kp0[hi * 8];
    kb.q[1] = *(const v4u*)&kp0[16 + hi * 8];
    s0 = __builtin_amdgcn_wmma_f32_16x16x32_bf16(false, qa0.v, false, kb.v, (short)0, s0, false, false);
    kb.q[0] = *(const v4u*)&kp0[32 + hi * 8];
    kb.q[1] = *(const v4u*)&kp0[48 + hi * 8];
    s0 = __builtin_amdgcn_wmma_f32_16x16x32_bf16(false, qa1.v, false, kb.v, (short)0, s0, false, false);
    const u16* kp1 = kp0 + (size_t)16 * KVD;                        // tile 1
    kb.q[0] = *(const v4u*)&kp1[hi * 8];
    kb.q[1] = *(const v4u*)&kp1[16 + hi * 8];
    s1 = __builtin_amdgcn_wmma_f32_16x16x32_bf16(false, qa0.v, false, kb.v, (short)0, s1, false, false);
    kb.q[0] = *(const v4u*)&kp1[32 + hi * 8];
    kb.q[1] = *(const v4u*)&kp1[48 + hi * 8];
    s1 = __builtin_amdgcn_wmma_f32_16x16x32_bf16(false, qa1.v, false, kb.v, (short)0, s1, false, false);

    // ---- causal mask + online softmax (rows live in 16-lane half-groups)
#pragma unroll
    for (int v = 0; v < 8; ++v) {
      const int qrow = q0 + v + hi * 8;
      float x0 = (sb + r      > qrow) ? -1e30f : s0[v] * scale;
      float x1 = (sb + 16 + r > qrow) ? -1e30f : s1[v] * scale;
      float mx = fmaxf(x0, x1);
#pragma unroll
      for (int off = 1; off < 16; off <<= 1) mx = fmaxf(mx, __shfl_xor(mx, off, 32));
      float nm = fmaxf(mrun[v], mx);
      float p0 = __expf(x0 - nm);
      float p1 = __expf(x1 - nm);
      float sum = p0 + p1;
#pragma unroll
      for (int off = 1; off < 16; off <<= 1) sum += __shfl_xor(sum, off, 32);
      float corr = __expf(mrun[v] - nm);
      mrun[v] = nm;
      lrun[v] = lrun[v] * corr + sum;
      o0[v] *= corr; o1[v] *= corr; o2[v] *= corr; o3[v] *= corr;
      // stage P (bf16) in LDS: [row m][k = s-offset in block]
      Pl[(v + hi * 8) * 32 + r]      = f2bf(p0);
      Pl[(v + hi * 8) * 32 + 16 + r] = f2bf(p1);
    }
    __syncthreads();

    // ---- O += P(16x32) * V(32x64): read P back in A-frag layout
    Frag pa;
    pa.q[0] = *(const v4u*)&Pl[r * 32 + hi * 8];
    pa.q[1] = *(const v4u*)&Pl[r * 32 + 16 + hi * 8];
    const u16* vp = Vt + (size_t)(kv * Dh + r) * Cdim + sb;   // row d, cols s
    Frag vb;
    vb.q[0] = *(const v4u*)&vp[hi * 8];
    vb.q[1] = *(const v4u*)&vp[16 + hi * 8];
    o0 = __builtin_amdgcn_wmma_f32_16x16x32_bf16(false, pa.v, false, vb.v, (short)0, o0, false, false);
    vp += (size_t)16 * Cdim;
    vb.q[0] = *(const v4u*)&vp[hi * 8];
    vb.q[1] = *(const v4u*)&vp[16 + hi * 8];
    o1 = __builtin_amdgcn_wmma_f32_16x16x32_bf16(false, pa.v, false, vb.v, (short)0, o1, false, false);
    vp += (size_t)16 * Cdim;
    vb.q[0] = *(const v4u*)&vp[hi * 8];
    vb.q[1] = *(const v4u*)&vp[16 + hi * 8];
    o2 = __builtin_amdgcn_wmma_f32_16x16x32_bf16(false, pa.v, false, vb.v, (short)0, o2, false, false);
    vp += (size_t)16 * Cdim;
    vb.q[0] = *(const v4u*)&vp[hi * 8];
    vb.q[1] = *(const v4u*)&vp[16 + hi * 8];
    o3 = __builtin_amdgcn_wmma_f32_16x16x32_bf16(false, pa.v, false, vb.v, (short)0, o3, false, false);
    __syncthreads();
  }

  // ---- normalize and emit bf16 hidden [C][H*D]
#pragma unroll
  for (int v = 0; v < 8; ++v) {
    float inv = 1.0f / lrun[v];
    const size_t base = (size_t)(q0 + v + hi * 8) * HD + h * Dh + r;
    Hid[base]      = f2bf(o0[v] * inv);
    Hid[base + 16] = f2bf(o1[v] * inv);
    Hid[base + 32] = f2bf(o2[v] * inv);
    Hid[base + 48] = f2bf(o3[v] * inv);
  }
}

// ---------------------------------------------------------------- launch
extern "C" void kernel_launch(void* const* d_in, const int* in_sizes, int n_in,
                              void* d_out, int out_size, void* d_ws, size_t ws_size,
                              hipStream_t stream) {
  (void)in_sizes; (void)n_in; (void)out_size; (void)ws_size;
  const float* x  = (const float*)d_in[0];
  const float* Wq = (const float*)d_in[1];
  const float* Wk = (const float*)d_in[2];
  const float* Wv = (const float*)d_in[3];
  const float* Wo = (const float*)d_in[4];
  float* out = (float*)d_out;

  const size_t nXE = (size_t)Cdim * Edim;     // 4194304
  const size_t nKV = (size_t)KVD  * Edim;     // 1048576
  u16* p = (u16*)d_ws;
  u16* xb  = p; p += nXE;
  u16* Wqb = p; p += nXE;
  u16* Wkb = p; p += nKV;
  u16* Wvb = p; p += nKV;
  u16* Wob = p; p += nXE;
  u16* Qb  = p; p += nXE;
  u16* Kb  = p; p += (size_t)Cdim * KVD;
  u16* Vb  = p; p += (size_t)Cdim * KVD;
  u16* Vtb = p; p += (size_t)Cdim * KVD;
  u16* Hb  = p; p += nXE;

  const int T = 256;
  f32_to_bf16_kernel<<<nXE / T, T, 0, stream>>>(x,  xb,  (int)nXE);
  f32_to_bf16_kernel<<<nXE / T, T, 0, stream>>>(Wq, Wqb, (int)nXE);
  f32_to_bf16_kernel<<<nKV / T, T, 0, stream>>>(Wk, Wkb, (int)nKV);
  f32_to_bf16_kernel<<<nKV / T, T, 0, stream>>>(Wv, Wvb, (int)nKV);
  f32_to_bf16_kernel<<<nXE / T, T, 0, stream>>>(Wo, Wob, (int)nXE);

  // projections: Q [C,HD], K [C,KVD], V [C,KVD]  (block tile 128x64)
  gemm_bf16_kernel<false><<<dim3(HD  / 64, Cdim / 128), T, 0, stream>>>(xb, Wqb, Qb, HD,  Edim);
  gemm_bf16_kernel<false><<<dim3(KVD / 64, Cdim / 128), T, 0, stream>>>(xb, Wkb, Kb, KVD, Edim);
  gemm_bf16_kernel<false><<<dim3(KVD / 64, Cdim / 128), T, 0, stream>>>(xb, Wvb, Vb, KVD, Edim);

  transpose_v_kernel<<<((size_t)Cdim * KVD) / T, T, 0, stream>>>(Vb, Vtb);

  gqa_attention_kernel<<<dim3(Cdim / 16, Hn), 32, 0, stream>>>(Qb, Kb, Vtb, Hb);

  // output projection -> fp32
  gemm_bf16_kernel<true><<<dim3(Edim / 64, Cdim / 128), T, 0, stream>>>(Hb, Wob, out, Edim, Edim);
}
